// Llama_11768210391064
// MI455X (gfx1250) — matrix operand order repaired and checked by reading, not compiled, not run on hardware
//
#include <hip/hip_runtime.h>
#include <math.h>

typedef __attribute__((ext_vector_type(16))) _Float16 v16h;
typedef __attribute__((ext_vector_type(8)))  float    v8f;

#define DMODEL  1024
#define HFF     4096
#define NHEAD   16
#define HEADDIM 64
#define NLAYER  4
#define VOCAB   32000
#define BATCH   2
#define SEQ     2048
#define MROWS   (BATCH*SEQ)   // 4096

union AF { v16h v; _Float16 e[16]; };

// Pack 4 floats -> 4 f16 in a uint2 (one ds_store_b64)
__device__ inline uint2 pack4(float a, float b, float c, float d) {
  union { _Float16 h[4]; uint2 u; } p;
  p.h[0] = (_Float16)a; p.h[1] = (_Float16)b; p.h[2] = (_Float16)c; p.h[3] = (_Float16)d;
  return p.u;
}

// Fragment layouts used here (contiguous k order, same for A and B):
// A-frag (16x32): lane = (m&15) + 16*((k>>4)&1), elem = k&15
// B-frag (32x16): lane = (n&15) + 16*((k>>4)&1), elem = k&15

// ---------------- embedding gather ----------------
__global__ void embed_kernel(const int* __restrict__ idx, const float* __restrict__ emb,
                             float* __restrict__ x) {
  int i = blockIdx.x * 256 + threadIdx.x;
  if (i >= MROWS * DMODEL) return;
  int row = i / DMODEL, c = i % DMODEL;
  x[i] = emb[(size_t)idx[row] * DMODEL + c];
}

// ---------------- RMSNorm (block per row) ----------------
__global__ void rmsnorm_kernel(const float* __restrict__ x, const float* __restrict__ w,
                               float* __restrict__ out, int Dd) {
  __shared__ float red[256];
  int row = blockIdx.x;
  const float* xr = x + (size_t)row * Dd;
  float s = 0.f;
  for (int c = threadIdx.x; c < Dd; c += 256) { float v = xr[c]; s += v * v; }
  red[threadIdx.x] = s; __syncthreads();
  for (int st = 128; st > 0; st >>= 1) {
    if (threadIdx.x < st) red[threadIdx.x] += red[threadIdx.x + st];
    __syncthreads();
  }
  float inv = rsqrtf(red[0] / (float)Dd + 1e-6f);
  for (int c = threadIdx.x; c < Dd; c += 256)
    out[(size_t)row * Dd + c] = xr[c] * inv * w[c];
}

// ---------------- tiled WMMA GEMM: C[M,N] = A[M,K] x B + optional residual ----------------
// Block tile 64x64, K-stage 64, 4 waves each owning a 32x32 sub-tile.
// LDS holds data pre-swizzled into WMMA fragment layout -> frag read = 1 v16h load (2x ds_load_b128).
template <bool TRANSB>
__global__ void gemm_kernel(const float* __restrict__ A, const float* __restrict__ B,
                            const float* __restrict__ res, float* __restrict__ C,
                            int M, int N, int K) {
  __shared__ __align__(32) _Float16 AsF[2][4][32][16];  // [kt][mt][lane][elem]
  __shared__ __align__(32) _Float16 BsF[2][4][32][16];  // [kt][nt][lane][elem]
  int tid = threadIdx.x, lane = tid & 31, wave = tid >> 5;
  int bm = blockIdx.y * 64, bn = blockIdx.x * 64;
  int wm = (wave >> 1) * 32, wn = (wave & 1) * 32;
  int mt0 = wm >> 4, nt0 = wn >> 4;
  v8f acc[2][2] = {};
  for (int k0 = 0; k0 < K; k0 += 64) {
    // A: 64 rows x 64 k, float4 along k -> one ds_store_b64 per quad
    for (int i = tid; i < 1024; i += 128) {
      int ml = i >> 4, kk = (i & 15) * 4;
      float4 a4 = *(const float4*)(A + (size_t)(bm + ml) * K + k0 + kk);
      int kt = kk >> 5, kl = kk & 31;
      int ln = (ml & 15) + 16 * (kl >> 4);
      int u2 = (kl & 15) >> 2;          // uint2 slot
      ((uint2*)&AsF[kt][ml >> 4][ln][0])[u2] = pack4(a4.x, a4.y, a4.z, a4.w);
    }
    if (TRANSB) {
      // B is NxK row-major: float4 along k -> one ds_store_b64 per quad
      for (int i = tid; i < 1024; i += 128) {
        int nl = i >> 4, kk = (i & 15) * 4;
        float4 b4 = *(const float4*)(B + (size_t)(bn + nl) * K + k0 + kk);
        int kt = kk >> 5, kl = kk & 31;
        int ln = (nl & 15) + 16 * (kl >> 4);
        ((uint2*)&BsF[kt][nl >> 4][ln][0])[(kl & 15) >> 2] = pack4(b4.x, b4.y, b4.z, b4.w);
      }
    } else {
      // B is KxN row-major: float4 along n -> 4 ds_store_b16
      for (int i = tid; i < 1024; i += 128) {
        int r = i >> 4, nn = (i & 15) * 4;
        float4 b4 = *(const float4*)(B + (size_t)(k0 + r) * N + bn + nn);
        int kt = r >> 5, kl = r & 31;
        int hl = 16 * (kl >> 4);
        int el = kl & 15;
        int nt = nn >> 4;
        BsF[kt][nt][((nn + 0) & 15) + hl][el] = (_Float16)b4.x;
        BsF[kt][nt][((nn + 1) & 15) + hl][el] = (_Float16)b4.y;
        BsF[kt][nt][((nn + 2) & 15) + hl][el] = (_Float16)b4.z;
        BsF[kt][nt][((nn + 3) & 15) + hl][el] = (_Float16)b4.w;
      }
    }
    __syncthreads();
    AF af[2][2], bf[2][2];
#pragma unroll
    for (int kt = 0; kt < 2; ++kt) {
      af[kt][0].v = *(const v16h*)&AsF[kt][mt0    ][lane][0];
      af[kt][1].v = *(const v16h*)&AsF[kt][mt0 + 1][lane][0];
      bf[kt][0].v = *(const v16h*)&BsF[kt][nt0    ][lane][0];
      bf[kt][1].v = *(const v16h*)&BsF[kt][nt0 + 1][lane][0];
    }
#pragma unroll
    for (int kt = 0; kt < 2; ++kt)
#pragma unroll
      for (int mt = 0; mt < 2; ++mt)
#pragma unroll
        for (int nt = 0; nt < 2; ++nt)
          acc[mt][nt] = __builtin_amdgcn_wmma_f32_16x16x32_f16(
              false, af[kt][mt].v, false, bf[kt][nt].v, (short)0, acc[mt][nt], false, false);
    __syncthreads();
  }
#pragma unroll
  for (int mt = 0; mt < 2; ++mt)
#pragma unroll
    for (int nt = 0; nt < 2; ++nt)
#pragma unroll
      for (int i = 0; i < 8; ++i) {
        int row = bm + wm + 16 * mt + i + ((lane < 16) ? 0 : 8);
        int col = bn + wn + 16 * nt + (lane & 15);
        size_t off = (size_t)row * N + col;
        float vv = acc[mt][nt][i];
        if (res) vv += res[off];
        C[off] = vv;
      }
}

// ---------------- RoPE (in place on q or k, (b,t,h,d) layout) ----------------
__global__ void rope_kernel(float* __restrict__ x) {
  int i = blockIdx.x * 256 + threadIdx.x;
  if (i >= MROWS * NHEAD * (HEADDIM / 2)) return;
  int p   = i & 31;
  int h   = (i >> 5) & (NHEAD - 1);
  int row = i >> 9;
  int t   = row & (SEQ - 1);
  float ang = (float)t * expf(-(2.0f * (float)p / (float)HEADDIM) * logf(10000.0f));
  float c = cosf(ang), s = sinf(ang);
  size_t base = (size_t)row * DMODEL + h * HEADDIM;
  float x1 = x[base + p], x2 = x[base + 32 + p];
  x[base + p]      = x1 * c - x2 * s;
  x[base + 32 + p] = x2 * c + x1 * s;
}

// ---------------- flash attention: block per (qblock64, head, batch), 4 waves ----------------
__global__ void attn_kernel(const float* __restrict__ Q, const float* __restrict__ K,
                            const float* __restrict__ Vv, float* __restrict__ O) {
  __shared__ __align__(32) _Float16 QF[2][4][32][16];  // A-frag layout [kt][mt][lane][elem]
  __shared__ __align__(32) _Float16 KF[2][4][32][16];  // B-frag layout (k=hd, n=key)
  __shared__ __align__(32) _Float16 VF[2][4][32][16];  // B-frag layout (k=key, n=hd)
  __shared__ __align__(32) _Float16 PF[2][4][32][16];  // A-frag layout (m=query, k=key)
  __shared__ float Ss[64][68];
  __shared__ float mstat[64], lstat[64], astat[64];
  int tid = threadIdx.x, lane = tid & 31, wave = tid >> 5;
  int qb = blockIdx.x, hh = blockIdx.y, bb = blockIdx.z;
  size_t rowbase = (size_t)bb * SEQ;
  // stage Q in A-frag layout
  for (int i = tid; i < 1024; i += 128) {
    int ml = i >> 4, dd = (i & 15) * 4;
    float4 q4 = *(const float4*)(Q + (rowbase + qb * 64 + ml) * DMODEL + hh * HEADDIM + dd);
    int kt = dd >> 5, kl = dd & 31;
    int ln = (ml & 15) + 16 * (kl >> 4);
    int u2 = (kl & 15) >> 2;
    ((uint2*)&QF[kt][ml >> 4][ln][0])[u2] = pack4(q4.x, q4.y, q4.z, q4.w);
  }
  if (tid < 64) { mstat[tid] = -3.0e38f; lstat[tid] = 0.f; }
  __syncthreads();
  AF qf[2];
  qf[0].v = *(const v16h*)&QF[0][wave][lane][0];
  qf[1].v = *(const v16h*)&QF[1][wave][lane][0];
  v8f oacc[4] = {};
  for (int kb = 0; kb <= qb; ++kb) {
    __syncthreads();
    // K block -> B-frag (k=hd d, n=key j): float4 along d = along k -> ds_store_b64
    for (int i = tid; i < 1024; i += 128) {
      int jl = i >> 4, dd = (i & 15) * 4;
      float4 k4 = *(const float4*)(K + (rowbase + kb * 64 + jl) * DMODEL + hh * HEADDIM + dd);
      int kt = dd >> 5, kl = dd & 31;
      int ln = (jl & 15) + 16 * (kl >> 4);
      ((uint2*)&KF[kt][jl >> 4][ln][0])[(kl & 15) >> 2] = pack4(k4.x, k4.y, k4.z, k4.w);
    }
    // V block -> B-frag (k=key j, n=hd d): float4 along d = along n -> 4 ds_store_b16
    for (int i = tid; i < 1024; i += 128) {
      int jl = i >> 4, dd = (i & 15) * 4;
      float4 v4 = *(const float4*)(Vv + (rowbase + kb * 64 + jl) * DMODEL + hh * HEADDIM + dd);
      int kt = jl >> 5, kl = jl & 31;
      int hl = 16 * (kl >> 4);
      int el = kl & 15;
      int nt = dd >> 4;
      VF[kt][nt][((dd + 0) & 15) + hl][el] = (_Float16)v4.x;
      VF[kt][nt][((dd + 1) & 15) + hl][el] = (_Float16)v4.y;
      VF[kt][nt][((dd + 2) & 15) + hl][el] = (_Float16)v4.z;
      VF[kt][nt][((dd + 3) & 15) + hl][el] = (_Float16)v4.w;
    }
    __syncthreads();
    // S strip = Q_strip x K^T
    v8f s[4] = {};
#pragma unroll
    for (int ks = 0; ks < 2; ++ks)
#pragma unroll
      for (int nt = 0; nt < 4; ++nt) {
        AF bf; bf.v = *(const v16h*)&KF[ks][nt][lane][0];
        s[nt] = __builtin_amdgcn_wmma_f32_16x16x32_f16(
            false, qf[ks].v, false, bf.v, (short)0, s[nt], false, false);
      }
#pragma unroll
    for (int nt = 0; nt < 4; ++nt)
#pragma unroll
      for (int i = 0; i < 8; ++i) {
        int r = 16 * wave + i + ((lane < 16) ? 0 : 8);
        Ss[r][16 * nt + (lane & 15)] = s[nt][i];
      }
    __syncthreads();
    if (tid < 64) {                    // online softmax, one row per thread, float4 LDS reads
      int r = tid;
      int lim = qb * 64 + r - kb * 64; // valid columns c <= lim
      float mold = mstat[r], bm = -3.0e38f;
      for (int c = 0; c < 64; c += 4) {
        float4 sv = *(const float4*)&Ss[r][c];
        if (c + 0 <= lim) bm = fmaxf(bm, sv.x * 0.125f);
        if (c + 1 <= lim) bm = fmaxf(bm, sv.y * 0.125f);
        if (c + 2 <= lim) bm = fmaxf(bm, sv.z * 0.125f);
        if (c + 3 <= lim) bm = fmaxf(bm, sv.w * 0.125f);
      }
      float mnew  = fmaxf(mold, bm);
      float alpha = expf(mold - mnew);
      float ls = 0.f;
      int mm = r & 15, mt = r >> 4;
      for (int c = 0; c < 64; c += 4) {
        float4 sv = *(const float4*)&Ss[r][c];
        float p0 = (c + 0 <= lim) ? expf(sv.x * 0.125f - mnew) : 0.f;
        float p1 = (c + 1 <= lim) ? expf(sv.y * 0.125f - mnew) : 0.f;
        float p2 = (c + 2 <= lim) ? expf(sv.z * 0.125f - mnew) : 0.f;
        float p3 = (c + 3 <= lim) ? expf(sv.w * 0.125f - mnew) : 0.f;
        ls += p0 + p1 + p2 + p3;
        int kt = c >> 5, kl = c & 31;
        int ln = mm + 16 * (kl >> 4);
        int u2 = (kl & 15) >> 2;
        ((uint2*)&PF[kt][mt][ln][0])[u2] = pack4(p0, p1, p2, p3);
      }
      lstat[r] = alpha * lstat[r] + ls;
      mstat[r] = mnew;
      astat[r] = alpha;
    }
    __syncthreads();
    float al[8];
#pragma unroll
    for (int i = 0; i < 8; ++i) al[i] = astat[16 * wave + i + ((lane < 16) ? 0 : 8)];
#pragma unroll
    for (int nt = 0; nt < 4; ++nt)
#pragma unroll
      for (int i = 0; i < 8; ++i) oacc[nt][i] *= al[i];
    AF pf[2];
    pf[0].v = *(const v16h*)&PF[0][wave][lane][0];
    pf[1].v = *(const v16h*)&PF[1][wave][lane][0];
#pragma unroll
    for (int ks = 0; ks < 2; ++ks)
#pragma unroll
      for (int nt = 0; nt < 4; ++nt) {
        AF bf; bf.v = *(const v16h*)&VF[ks][nt][lane][0];
        oacc[nt] = __builtin_amdgcn_wmma_f32_16x16x32_f16(
            false, pf[ks].v, false, bf.v, (short)0, oacc[nt], false, false);
      }
  }
#pragma unroll
  for (int i = 0; i < 8; ++i) {
    int r = 16 * wave + i + ((lane < 16) ? 0 : 8);
    float invl = 1.0f / lstat[r];
#pragma unroll
    for (int nt = 0; nt < 4; ++nt)
      O[(rowbase + qb * 64 + r) * DMODEL + hh * HEADDIM + 16 * nt + (lane & 15)] =
          oacc[nt][i] * invl;
  }
}

// ---------------- SiLU(gate) * up, in place on g ----------------
__global__ void silu_mul_kernel(float* __restrict__ g, const float* __restrict__ u, size_t n) {
  size_t i = (size_t)blockIdx.x * 256 + threadIdx.x;
  if (i >= n) return;
  float gv = g[i];
  g[i] = (gv / (1.0f + expf(-gv))) * u[i];
}

// ---------------- per-row NLL over VOCAB logits ----------------
__global__ void nll_kernel(const float* __restrict__ logits, const int* __restrict__ tgt,
                           float* __restrict__ nll) {
  __shared__ float red[256];
  int row = blockIdx.x;
  const float* lr = logits + (size_t)row * VOCAB;
  float mx = -3.0e38f;
  for (int c = threadIdx.x; c < VOCAB; c += 256) mx = fmaxf(mx, lr[c]);
  red[threadIdx.x] = mx; __syncthreads();
  for (int st = 128; st > 0; st >>= 1) {
    if (threadIdx.x < st) red[threadIdx.x] = fmaxf(red[threadIdx.x], red[threadIdx.x + st]);
    __syncthreads();
  }
  mx = red[0]; __syncthreads();
  float s = 0.f;
  for (int c = threadIdx.x; c < VOCAB; c += 256) s += expf(lr[c] - mx);
  red[threadIdx.x] = s; __syncthreads();
  for (int st = 128; st > 0; st >>= 1) {
    if (threadIdx.x < st) red[threadIdx.x] += red[threadIdx.x + st];
    __syncthreads();
  }
  if (threadIdx.x == 0) nll[row] = -(lr[tgt[row]] - mx - logf(red[0]));
}

__global__ void loss_kernel(const float* __restrict__ nll, float* __restrict__ out) {
  __shared__ float red[256];
  float s = 0.f;
  for (int i = threadIdx.x; i < MROWS; i += 256) s += nll[i];
  red[threadIdx.x] = s; __syncthreads();
  for (int st = 128; st > 0; st >>= 1) {
    if (threadIdx.x < st) red[threadIdx.x] += red[threadIdx.x + st];
    __syncthreads();
  }
  if (threadIdx.x == 0) out[0] = red[0] / (float)MROWS;
}

extern "C" void kernel_launch(void* const* d_in, const int* in_sizes, int n_in,
                              void* d_out, int out_size, void* d_ws, size_t ws_size,
                              hipStream_t stream) {
  (void)in_sizes; (void)n_in; (void)out_size; (void)ws_size;
  const int*   idx      = (const int*)  d_in[0];
  const int*   targets  = (const int*)  d_in[1];
  const float* embed    = (const float*)d_in[2];
  const float* wq       = (const float*)d_in[3];
  const float* wk       = (const float*)d_in[4];
  const float* wv       = (const float*)d_in[5];
  const float* wo       = (const float*)d_in[6];
  const float* wgate    = (const float*)d_in[7];
  const float* wup      = (const float*)d_in[8];
  const float* wdown    = (const float*)d_in[9];
  const float* anorm    = (const float*)d_in[10];
  const float* mnorm    = (const float*)d_in[11];
  const float* fnorm    = (const float*)d_in[12];
  const float* lm_head  = (const float*)d_in[13];
  float* out = (float*)d_out;
  float* ws  = (float*)d_ws;

  const size_t XSZ = (size_t)MROWS * DMODEL;
  const size_t FSZ = (size_t)MROWS * HFF;
  float* x  = ws;
  float* h  = x  + XSZ;
  float* q  = h  + XSZ;
  float* k  = q  + XSZ;
  float* v  = k  + XSZ;
  float* ao = v  + XSZ;
  float* g  = ao + XSZ;
  float* u  = g  + FSZ;
  float* nl = u  + FSZ;

  embed_kernel<<<(MROWS * DMODEL + 255) / 256, 256, 0, stream>>>(idx, embed, x);

  dim3 gq(DMODEL / 64, MROWS / 64);
  dim3 gf(HFF / 64,    MROWS / 64);
  int ropeBlocks = (MROWS * NHEAD * (HEADDIM / 2) + 255) / 256;

  for (int l = 0; l < NLAYER; ++l) {
    const float* wql = wq    + (size_t)l * DMODEL * DMODEL;
    const float* wkl = wk    + (size_t)l * DMODEL * DMODEL;
    const float* wvl = wv    + (size_t)l * DMODEL * DMODEL;
    const float* wol = wo    + (size_t)l * DMODEL * DMODEL;
    const float* wgl = wgate + (size_t)l * DMODEL * HFF;
    const float* wul = wup   + (size_t)l * DMODEL * HFF;
    const float* wdl = wdown + (size_t)l * HFF * DMODEL;

    rmsnorm_kernel<<<MROWS, 256, 0, stream>>>(x, anorm + (size_t)l * DMODEL, h, DMODEL);
    gemm_kernel<false><<<gq, 128, 0, stream>>>(h, wql, nullptr, q, MROWS, DMODEL, DMODEL);
    gemm_kernel<false><<<gq, 128, 0, stream>>>(h, wkl, nullptr, k, MROWS, DMODEL, DMODEL);
    gemm_kernel<false><<<gq, 128, 0, stream>>>(h, wvl, nullptr, v, MROWS, DMODEL, DMODEL);
    rope_kernel<<<ropeBlocks, 256, 0, stream>>>(q);
    rope_kernel<<<ropeBlocks, 256, 0, stream>>>(k);
    attn_kernel<<<dim3(SEQ / 64, NHEAD, BATCH), 128, 0, stream>>>(q, k, v, ao);
    gemm_kernel<false><<<gq, 128, 0, stream>>>(ao, wol, x, x, MROWS, DMODEL, DMODEL);

    rmsnorm_kernel<<<MROWS, 256, 0, stream>>>(x, mnorm + (size_t)l * DMODEL, h, DMODEL);
    gemm_kernel<false><<<gf, 128, 0, stream>>>(h, wgl, nullptr, g, MROWS, HFF, DMODEL);
    gemm_kernel<false><<<gf, 128, 0, stream>>>(h, wul, nullptr, u, MROWS, HFF, DMODEL);
    silu_mul_kernel<<<(unsigned)((FSZ + 255) / 256), 256, 0, stream>>>(g, u, FSZ);
    gemm_kernel<false><<<gq, 128, 0, stream>>>(g, wdl, x, x, MROWS, DMODEL, HFF);
  }

  rmsnorm_kernel<<<MROWS, 256, 0, stream>>>(x, fnorm, h, DMODEL);
  gemm_kernel<true><<<dim3(VOCAB / 64, MROWS / 64), 128, 0, stream>>>(
      h, lm_head, nullptr, out, MROWS, VOCAB, DMODEL);
  nll_kernel<<<MROWS, 256, 0, stream>>>(out, targets, nl);
  loss_kernel<<<1, 256, 0, stream>>>(nl, out + (size_t)MROWS * VOCAB);
}
